// CTSSF_70411693850698
// MI455X (gfx1250) — matrix operand order repaired and checked by reading, not compiled, not run on hardware
//
#include <hip/hip_runtime.h>
#include <hip/hip_bf16.h>
#include <stdint.h>

#define BATCH   8192
#define N_CAL   8192
#define IN_DIM  60
#define HID     128
#define LAT     64
#define NTILES  (N_CAL / 16)   // 512 column tiles of 16

typedef __attribute__((ext_vector_type(16))) __bf16 v16bf;
typedef __attribute__((ext_vector_type(8)))  float  v8f;

// bf16 <-> f32 via bit ops (RNE) — avoids relying on __bf16 scalar arithmetic.
__device__ __forceinline__ unsigned short f2bf(float f) {
  unsigned u = __float_as_uint(f);
  u += 0x7FFFu + ((u >> 16) & 1u);
  return (unsigned short)(u >> 16);
}
__device__ __forceinline__ float bf2f(unsigned short h) {
  return __uint_as_float(((unsigned)h) << 16);
}

// ---------------------------------------------------------------------------
// Stable rank of each calibration score (argsort replacement, O(N^2) but the
// 32KB score array is L0/L2 resident). Also scatters sorted_scores.
// ---------------------------------------------------------------------------
__global__ void rank_kernel(const float* __restrict__ scores,
                            int* __restrict__ rank,
                            float* __restrict__ sorted_scores) {
  int j = blockIdx.x * blockDim.x + threadIdx.x;
  if (j >= N_CAL) return;
  float sj = scores[j];
  int rk = 0;
  for (int k = 0; k < N_CAL; ++k) {
    float sk = scores[k];
    rk += (sk < sj) || (sk == sj && k < j);   // stable tie-break like argsort
  }
  rank[j] = rk;
  sorted_scores[rk] = sj;
}

// ---------------------------------------------------------------------------
// Normalize calibration latents and scatter rows into sorted-score order as
// bf16 (WMMA A-operand staging).
// ---------------------------------------------------------------------------
__global__ void cnorm_kernel(const float* __restrict__ cal,
                             const int* __restrict__ rank,
                             unsigned short* __restrict__ cnSb) {
  int j = blockIdx.x * blockDim.x + threadIdx.x;
  if (j >= N_CAL) return;
  const float* row = cal + (size_t)j * LAT;
  float ss = 0.f;
  for (int d = 0; d < LAT; ++d) { float v = row[d]; ss = fmaf(v, v, ss); }
  float inv = 1.f / (sqrtf(ss) + 1e-8f);
  unsigned short* out = cnSb + (size_t)rank[j] * LAT;
  for (int d = 0; d < LAT; ++d) out[d] = f2bf(row[d] * inv);
}

// ---------------------------------------------------------------------------
// Encoder MLP + LayerNorm + L2-normalize; one row per 64-thread block.
// ~0.5 GFLOP total: latency-bound, VALU is fine.
// ---------------------------------------------------------------------------
__global__ void encode_kernel(const float* __restrict__ x,
                              const float* __restrict__ W1, const float* __restrict__ b1,
                              const float* __restrict__ W2, const float* __restrict__ b2,
                              const float* __restrict__ W3, const float* __restrict__ b3,
                              const float* __restrict__ lnw, const float* __restrict__ lnb,
                              unsigned short* __restrict__ qnb) {
  __shared__ float xs[IN_DIM];
  __shared__ float h1[HID];
  __shared__ float h2[HID];
  __shared__ float red[LAT];
  const int row = blockIdx.x;
  const int tid = threadIdx.x;            // 0..63

  if (tid < IN_DIM) xs[tid] = x[(size_t)row * IN_DIM + tid];
  __syncthreads();

  for (int c = tid; c < HID; c += 64) {
    float acc = b1[c];
    for (int d = 0; d < IN_DIM; ++d) acc = fmaf(xs[d], W1[d * HID + c], acc);
    h1[c] = fmaxf(acc, 0.f);
  }
  __syncthreads();
  for (int c = tid; c < HID; c += 64) {
    float acc = b2[c];
    for (int d = 0; d < HID; ++d) acc = fmaf(h1[d], W2[d * HID + c], acc);
    h2[c] = fmaxf(acc, 0.f);
  }
  __syncthreads();

  float z = b3[tid];
  for (int d = 0; d < HID; ++d) z = fmaf(h2[d], W3[d * LAT + tid], z);

  // mean
  red[tid] = z; __syncthreads();
  for (int s = 32; s > 0; s >>= 1) { if (tid < s) red[tid] += red[tid + s]; __syncthreads(); }
  float mu = red[0] * (1.f / LAT); __syncthreads();
  // variance (population)
  float zc = z - mu;
  red[tid] = zc * zc; __syncthreads();
  for (int s = 32; s > 0; s >>= 1) { if (tid < s) red[tid] += red[tid + s]; __syncthreads(); }
  float var = red[0] * (1.f / LAT); __syncthreads();

  float y = zc * rsqrtf(var + 1e-5f) * lnw[tid] + lnb[tid];

  // L2 norm
  red[tid] = y * y; __syncthreads();
  for (int s = 32; s > 0; s >>= 1) { if (tid < s) red[tid] += red[tid + s]; __syncthreads(); }
  float inv = 1.f / (sqrtf(red[0]) + 1e-8f);
  qnb[(size_t)row * LAT + tid] = f2bf(y * inv);
}

// ---------------------------------------------------------------------------
// Fused attention-softmax-quantile. One block = 16 query rows, 4 waves.
//
// Orientation: D = A x B with A = cal tile (M = calibration index) and
// B = qn^T (N = query index). With the 16x16 f32 D layout (VGPR k: lanes
// 0-15 -> M=k, lanes 16-31 -> M=k+8), the per-query softmax partial sum is a
// reduction over M = acc[0..7] *within one lane* + its xor-16 partner:
// 8 v_exp + 7 in-lane adds + ONE shuffle per tile (vs 32 ds_bpermute before).
//
// A tiles are double-buffered so next-tile global_load_b128s overlap the
// exp/reduce epilogue. Row totals Z accumulate in registers -> one ds_add_f32
// per wave at the end.
// ---------------------------------------------------------------------------
union Frag { v16bf v; uint4 u[2]; };

__device__ __forceinline__ void loadA(Frag& a0, Frag& a1,
                                      const unsigned short* __restrict__ cnSb,
                                      int c0, int l, int hi) {
  const unsigned short* pa = cnSb + (size_t)(c0 + l) * LAT + hi * 8;
  a0.u[0] = *(const uint4*)(pa);
  a0.u[1] = *(const uint4*)(pa + 16);
  a1.u[0] = *(const uint4*)(pa + 32);
  a1.u[1] = *(const uint4*)(pa + 48);
}

__global__ __launch_bounds__(128) void attn_kernel(
    const unsigned short* __restrict__ qnb,
    const unsigned short* __restrict__ cnSb,
    const float* __restrict__ sorted_scores,
    const float* __restrict__ pred,
    float* __restrict__ out) {
  __shared__ float tileSum[16][NTILES + 1];   // +1 pad: lane stride 513 -> conflict-free
  __shared__ float Zrow[16];

  const int r0   = blockIdx.x * 16;
  const int tid  = threadIdx.x;
  const int lane = tid & 31;
  const int wave = tid >> 5;
  const int l    = lane & 15;
  const int hi   = lane >> 4;

  if (tid < 16) Zrow[tid] = 0.f;
  __syncthreads();

  // B fragment: qn^T for the 16 query rows (fixed for the whole block).
  // B layout matches A: lane holds N=lane&15, K kb..kb+7 and kb+16..kb+23
  // with kb = 8*hi, per 32-wide K half.
  Frag b0, b1;
  {
    const unsigned short* pb = qnb + (size_t)(r0 + l) * LAT + hi * 8;
    b0.u[0] = *(const uint4*)(pb);
    b0.u[1] = *(const uint4*)(pb + 16);
    b1.u[0] = *(const uint4*)(pb + 32);
    b1.u[1] = *(const uint4*)(pb + 48);
  }

  // A fragment (ISA 7.12.2, 16-bit A 16x32): lane holds row M=l, K kb..kb+7
  // and kb+16..kb+23 with kb = 8*hi; two halves cover K=0..63.
  Frag a0, a1;
  loadA(a0, a1, cnSb, wave * 16, l, hi);

  float zpart = 0.f;
  for (int t = wave; t < NTILES; t += 4) {
    // Prefetch next tile's A operand (clamped to a valid address; the
    // redundant last-iteration load is discarded).
    Frag n0, n1;
    const int tn = (t + 4 < NTILES) ? (t + 4) : wave;
    loadA(n0, n1, cnSb, tn * 16, l, hi);

    v8f acc = {0.f, 0.f, 0.f, 0.f, 0.f, 0.f, 0.f, 0.f};
    acc = __builtin_amdgcn_wmma_f32_16x16x32_bf16(false, a0.v, false, b0.v,
                                                  (short)0, acc, false, false);
    acc = __builtin_amdgcn_wmma_f32_16x16x32_bf16(false, a1.v, false, b1.v,
                                                  (short)0, acc, false, false);

    // acc[k] = S[cal = t*16 + k + 8*hi][query = r0 + l]; scores in [-1,1]
    // (unit vectors, temp=1) so exp needs no max subtraction.
    float e0 = __expf(acc[0]), e1 = __expf(acc[1]);
    float e2 = __expf(acc[2]), e3 = __expf(acc[3]);
    float e4 = __expf(acc[4]), e5 = __expf(acc[5]);
    float e6 = __expf(acc[6]), e7 = __expf(acc[7]);
    float s01 = e0 + e1, s23 = e2 + e3, s45 = e4 + e5, s67 = e6 + e7;
    float local = (s01 + s23) + (s45 + s67);        // 8 cal rows of this half
    float tot = local + __shfl_xor(local, 16, 32);  // + partner half (1 DS op)

    if (hi == 0) tileSum[l][t] = tot;
    zpart += tot;                                   // both halves equal post-xor

    a0 = n0; a1 = n1;
  }
  if (hi == 0) atomicAdd(&Zrow[l], zpart);          // ds_add_f32, once per wave
  __syncthreads();

  if (tid < 16) {
    const int r = r0 + tid;
    const float target = 0.9f * Zrow[tid];          // 1 - ALPHA, unnormalized

    float cum = 0.f;
    int cnt = 0, tstar = -1;
    for (int t = 0; t < NTILES; ++t) {
      float s = tileSum[tid][t];
      if (cum + s < target) { cum += s; cnt += 16; }
      else { tstar = t; break; }
    }

    int idx;
    if (tstar < 0) {
      idx = N_CAL - 1;
    } else {
      // Refine inside the crossing tile: 16 VALU dot products (K=64).
      const unsigned short* qrow = qnb + (size_t)r * LAT;
      for (int n = 0; n < 16; ++n) {
        const unsigned short* crow = cnSb + (size_t)(tstar * 16 + n) * LAT;
        float s = 0.f;
        for (int d = 0; d < LAT; ++d) s = fmaf(bf2f(qrow[d]), bf2f(crow[d]), s);
        float e = __expf(s);
        if (cum + e < target) { cum += e; cnt += 1; }
        else break;
      }
      idx = cnt;
    }
    if (idx > N_CAL - 1) idx = N_CAL - 1;

    float w = sorted_scores[idx];
    w = fminf(fmaxf(w, 0.01f), 0.2f);
    float p = pred[r];
    out[r]         = p - w;
    out[BATCH + r] = p + w;
  }
}

// ---------------------------------------------------------------------------
extern "C" void kernel_launch(void* const* d_in, const int* in_sizes, int n_in,
                              void* d_out, int out_size, void* d_ws, size_t ws_size,
                              hipStream_t stream) {
  (void)in_sizes; (void)n_in; (void)out_size; (void)ws_size;
  const float* features    = (const float*)d_in[0];
  const float* predictions = (const float*)d_in[1];
  const float* cal_latents = (const float*)d_in[2];
  const float* cal_scores  = (const float*)d_in[3];
  const float* W1  = (const float*)d_in[4];
  const float* b1  = (const float*)d_in[5];
  const float* W2  = (const float*)d_in[6];
  const float* b2  = (const float*)d_in[7];
  const float* W3  = (const float*)d_in[8];
  const float* b3  = (const float*)d_in[9];
  const float* lnw = (const float*)d_in[10];
  const float* lnb = (const float*)d_in[11];
  float* out = (float*)d_out;

  // Workspace: qnb (1MB bf16) | cnSb (1MB bf16) | sorted_scores (32KB) | rank (32KB)
  char* ws = (char*)d_ws;
  unsigned short* qnb  = (unsigned short*)ws;
  unsigned short* cnSb = (unsigned short*)(ws + (size_t)BATCH * LAT * 2);
  float* sorted_scores = (float*)(ws + 2 * (size_t)BATCH * LAT * 2);
  int*   rank          = (int*)  (ws + 2 * (size_t)BATCH * LAT * 2 + (size_t)N_CAL * 4);

  rank_kernel  <<<N_CAL / 256, 256, 0, stream>>>(cal_scores, rank, sorted_scores);
  cnorm_kernel <<<N_CAL / 256, 256, 0, stream>>>(cal_latents, rank, cnSb);
  encode_kernel<<<BATCH, 64, 0, stream>>>(features, W1, b1, W2, b2, W3, b3, lnw, lnb, qnb);
  attn_kernel  <<<BATCH / 16, 128, 0, stream>>>(qnb, cnSb, sorted_scores, predictions, out);
}
